// EllipticGNN_60988535603849
// MI455X (gfx1250) — matrix operand order repaired and checked by reading, not compiled, not run on hardware
//
#include <hip/hip_runtime.h>
#include <hip/hip_bf16.h>

typedef __attribute__((ext_vector_type(16))) _Float16 v16h;
typedef __attribute__((ext_vector_type(8)))  _Float16 v8h;
typedef __attribute__((ext_vector_type(8)))  float    v8f;

#define F_IN  165
#define HIDD  128
#define EPSBN 1e-5f

__device__ __forceinline__ unsigned fenc(float f) {
  unsigned b = __float_as_uint(f);
  return (b & 0x80000000u) ? ~b : (b | 0x80000000u);
}
__device__ __forceinline__ float fdec(unsigned k) {
  return (k & 0x80000000u) ? __uint_as_float(k ^ 0x80000000u) : __uint_as_float(~k);
}
__device__ __forceinline__ float eluf(float x)  { return x > 0.f ? x : expm1f(x); }
__device__ __forceinline__ float lrelu(float x) { return x > 0.f ? x : 0.2f * x; }

// ---------------- zero ----------------
__global__ void k_zero(unsigned* __restrict__ p, long total) {
  long i = (long)blockIdx.x * blockDim.x + threadIdx.x;
  long s = (long)gridDim.x * blockDim.x;
  for (; i < total; i += s) p[i] = 0u;
}

// ---- weight convert f32[K][Nsrc] -> f16 TRANSPOSED dst[Npad][Kp] (zero padded) ----
__global__ void k_convert_wT(const float* __restrict__ src, _Float16* __restrict__ dst,
                             int K, int Nsrc, int Kp, int Npad) {
  int i = blockIdx.x * blockDim.x + threadIdx.x;
  int total = Kp * Npad;
  if (i >= total) return;
  int c = i / Kp, r = i % Kp;              // dst[c][r]
  float v = (r < K && c < Nsrc) ? src[r * Nsrc + c] : 0.f;
  dst[i] = (_Float16)v;
}

// ------------- generic WMMA GEMM: C[M,ldc] (+)= act(A[M,K] * Bt[Nout][Kp] + bias) -------------
// block = 256 threads = 8 waves; M-tile 128; wave w: rows [16w,16w+16) x NTILES 16-col tiles
template <int NTILES>
__global__ __launch_bounds__(256)
void k_gemm_wmma(const float* __restrict__ A, const _Float16* __restrict__ Bt,
                 float* __restrict__ C, const float* __restrict__ bias,
                 int M, int K, int Kp, int nstore, int ldc, int accum, int act) {
  constexpr int NOUT = NTILES * 16;
  __shared__ __align__(16) _Float16 As[128][40];     // [row][k], 80B stride (16B aligned)
  __shared__ __align__(16) _Float16 Bs[NOUT][40];    // [col][k], 80B stride (16B aligned)

  const int tid  = threadIdx.x;
  const int wave = tid >> 5;
  const int lane = tid & 31;
  const int m0   = blockIdx.x * 128;

  v8f accv[NTILES];
  #pragma unroll
  for (int t = 0; t < NTILES; ++t)
    #pragma unroll
    for (int e = 0; e < 8; ++e) accv[t][e] = 0.f;

  const int arow = wave * 16 + (lane & 15);
  const int kb   = (lane < 16) ? 0 : 8;    // A K-base
  const int kbb  = (lane < 16) ? 0 : 8;    // B K-base
  const int bcol = lane & 15;

  for (int k0 = 0; k0 < Kp; k0 += 32) {
    // speculative prefetch of next A chunk (gfx1250 global_prefetch_b8)
    if (k0 + 32 < Kp) {
      int pr = m0 + (tid >> 1);
      if (pr < M) __builtin_prefetch(A + (size_t)pr * K + k0 + 32, 0, 1);
    }
    // stage A tile: coalesced f32 global read -> f16 LDS
    for (int i = tid; i < 128 * 32; i += 256) {
      int r = i >> 5, c = i & 31;
      int gr = m0 + r, gc = k0 + c;
      float v = (gr < M && gc < K) ? A[(size_t)gr * K + gc] : 0.f;
      As[r][c] = (_Float16)v;
    }
    // stage B tile from pre-transposed weights: coalesced read, sequential LDS write
    for (int i = tid; i < NOUT * 32; i += 256) {
      int c = i >> 5, r = i & 31;
      Bs[c][r] = Bt[(size_t)c * Kp + k0 + r];
    }
    __syncthreads();

    // A fragment: two contiguous 16B LDS loads (ds_load_b128)
    v8h a0 = *(const v8h*)&As[arow][kb];
    v8h a1 = *(const v8h*)&As[arow][kb + 16];
    v16h afrag = __builtin_shufflevector(a0, a1, 0, 1, 2, 3, 4, 5, 6, 7,
                                         8, 9, 10, 11, 12, 13, 14, 15);

    // preload ALL B fragments (loads overlap; single dscnt drain), then
    // issue the independent WMMAs back-to-back to keep the matrix pipe busy
    v16h bfrag[NTILES];
    #pragma unroll
    for (int t = 0; t < NTILES; ++t) {
      const _Float16* bp = &Bs[t * 16 + bcol][kbb];
      v8h b0 = *(const v8h*)bp;
      v8h b1 = *(const v8h*)(bp + 16);
      bfrag[t] = __builtin_shufflevector(b0, b1, 0, 1, 2, 3, 4, 5, 6, 7,
                                         8, 9, 10, 11, 12, 13, 14, 15);
    }
    #pragma unroll
    for (int t = 0; t < NTILES; ++t) {
      accv[t] = __builtin_amdgcn_wmma_f32_16x16x32_f16(
          false, afrag, false, bfrag[t], (short)0, accv[t], false, false);
    }
    __syncthreads();
  }

  // epilogue: bias + optional accumulate + optional ELU
  const int nl   = lane & 15;
  const int mloc = (lane < 16) ? 0 : 8;
  #pragma unroll
  for (int t = 0; t < NTILES; ++t) {
    #pragma unroll
    for (int v = 0; v < 8; ++v) {
      int gm = m0 + wave * 16 + mloc + v;
      int gn = t * 16 + nl;
      if (gm < M && gn < nstore) {
        float val = accv[t][v];
        if (bias) val += bias[gn];
        size_t idx = (size_t)gm * ldc + gn;
        if (accum) val += C[idx];
        if (act) val = eluf(val);
        C[idx] = val;
      }
    }
  }
}

// ---------------- graph kernels ----------------
__global__ void k_degree(const int* __restrict__ dst, int E, int* __restrict__ cnt) {
  int i = blockIdx.x * blockDim.x + threadIdx.x;
  int s = gridDim.x * blockDim.x;
  for (; i < E; i += s) atomicAdd(&cnt[dst[i]], 1);
}

__global__ void k_dinv(const int* __restrict__ cnt, float* __restrict__ dinv, int n) {
  int i = blockIdx.x * blockDim.x + threadIdx.x;
  if (i < n) dinv[i] = rsqrtf((float)(cnt[i] + 1));   // +1 self loop, always > 0
}

// wave-per-edge: h1[dst] += hw[src] * dinv[s]*dinv[d]
__global__ void k_gcn_scatter(const float* __restrict__ hw, const int* __restrict__ src,
                              const int* __restrict__ dst, const float* __restrict__ dinv,
                              float* __restrict__ h1, int E) {
  int gw = (blockIdx.x * blockDim.x + threadIdx.x) >> 5;
  int nw = (gridDim.x * blockDim.x) >> 5;
  int lane = threadIdx.x & 31;
  for (int e = gw; e < E; e += nw) {
    int s = src[e], d = dst[e];
    float nrm = dinv[s] * dinv[d];
    const float* hs = hw + (size_t)s * HIDD;
    float* hd = h1 + (size_t)d * HIDD;
    #pragma unroll
    for (int c = 0; c < 4; ++c)
      atomicAdd(&hd[lane + c * 32], hs[lane + c * 32] * nrm);
  }
}

// h1 = elu(bn1(h1 + hw*dinv^2 + b_gcn))
__global__ void k_gcn_finish(float* __restrict__ h1, const float* __restrict__ hw,
                             const float* __restrict__ dinv, const float* __restrict__ bias,
                             const float* __restrict__ g, const float* __restrict__ b,
                             const float* __restrict__ m, const float* __restrict__ v,
                             int total) {
  int i = blockIdx.x * blockDim.x + threadIdx.x;
  if (i >= total) return;
  int c = i & (HIDD - 1);
  int node = i >> 7;
  float di = dinv[node];
  float val = h1[i] + hw[i] * di * di + bias[c];
  val = (val - m[c]) * (g[c] * rsqrtf(v[c] + EPSBN)) + b[c];
  h1[i] = eluf(val);
}

// a_src/a_dst per (node, head)
__global__ void k_gat_att(const float* __restrict__ g, const float* __restrict__ att_s,
                          const float* __restrict__ att_d, float* __restrict__ as,
                          float* __restrict__ ad, int n) {
  int t = blockIdx.x * blockDim.x + threadIdx.x;
  if (t >= n * 2) return;
  int head = t & 1, i = t >> 1;
  const float* gp = g + (size_t)i * HIDD + head * 64;
  float s = 0.f, d = 0.f;
  for (int c = 0; c < 64; ++c) {
    s += gp[c] * att_s[head * 64 + c];
    d += gp[c] * att_d[head * 64 + c];
  }
  as[t] = s; ad[t] = d;
}

__global__ void k_gat_edge_max(const int* __restrict__ src, const int* __restrict__ dst,
                               const float* __restrict__ as, const float* __restrict__ ad,
                               unsigned* __restrict__ mxu, int E) {
  int e = blockIdx.x * blockDim.x + threadIdx.x;
  int s0 = gridDim.x * blockDim.x;
  for (; e < E; e += s0) {
    int s = src[e], d = dst[e];
    #pragma unroll
    for (int h = 0; h < 2; ++h)
      atomicMax(&mxu[d * 2 + h], fenc(lrelu(as[s * 2 + h] + ad[d * 2 + h])));
  }
}

__global__ void k_gat_self_max(const float* __restrict__ as, const float* __restrict__ ad,
                               unsigned* __restrict__ mxu, int n2) {
  int t = blockIdx.x * blockDim.x + threadIdx.x;
  if (t >= n2) return;
  unsigned e = fenc(lrelu(as[t] + ad[t]));
  if (e > mxu[t]) mxu[t] = e;
}

__global__ void k_gat_edge_denom(const int* __restrict__ src, const int* __restrict__ dst,
                                 const float* __restrict__ as, const float* __restrict__ ad,
                                 const unsigned* __restrict__ mxu, float* __restrict__ den,
                                 int E) {
  int e = blockIdx.x * blockDim.x + threadIdx.x;
  int s0 = gridDim.x * blockDim.x;
  for (; e < E; e += s0) {
    int s = src[e], d = dst[e];
    #pragma unroll
    for (int h = 0; h < 2; ++h) {
      float v = lrelu(as[s * 2 + h] + ad[d * 2 + h]);
      atomicAdd(&den[d * 2 + h], __expf(v - fdec(mxu[d * 2 + h])));
    }
  }
}

__global__ void k_gat_self_denom(const float* __restrict__ as, const float* __restrict__ ad,
                                 const unsigned* __restrict__ mxu, float* __restrict__ den,
                                 int n2) {
  int t = blockIdx.x * blockDim.x + threadIdx.x;
  if (t >= n2) return;
  float v = lrelu(as[t] + ad[t]);
  den[t] += __expf(v - fdec(mxu[t]));
}

// wave-per-edge: h2[dst] += alpha * g[src]
__global__ void k_gat_edge_scatter(const int* __restrict__ src, const int* __restrict__ dst,
                                   const float* __restrict__ as, const float* __restrict__ ad,
                                   const unsigned* __restrict__ mxu, const float* __restrict__ den,
                                   const float* __restrict__ g, float* __restrict__ h2, int E) {
  int gw = (blockIdx.x * blockDim.x + threadIdx.x) >> 5;
  int nw = (gridDim.x * blockDim.x) >> 5;
  int lane = threadIdx.x & 31;
  for (int e = gw; e < E; e += nw) {
    int s = src[e], d = dst[e];
    float al[2];
    #pragma unroll
    for (int h = 0; h < 2; ++h) {
      float v = lrelu(as[s * 2 + h] + ad[d * 2 + h]);
      al[h] = __expf(v - fdec(mxu[d * 2 + h])) / (den[d * 2 + h] + 1e-16f);
    }
    const float* gs = g + (size_t)s * HIDD;
    float* hd = h2 + (size_t)d * HIDD;
    #pragma unroll
    for (int j = 0; j < 4; ++j) {
      int c = lane + j * 32;
      atomicAdd(&hd[c], al[c >> 6] * gs[c]);
    }
  }
}

__global__ void k_gat_self_scatter(const float* __restrict__ as, const float* __restrict__ ad,
                                   const unsigned* __restrict__ mxu, const float* __restrict__ den,
                                   const float* __restrict__ g, float* __restrict__ h2, int total) {
  int i = blockIdx.x * blockDim.x + threadIdx.x;
  if (i >= total) return;
  int c = i & (HIDD - 1);
  int node = i >> 7;
  int t = node * 2 + (c >> 6);
  float v = lrelu(as[t] + ad[t]);
  float al = __expf(v - fdec(mxu[t])) / (den[t] + 1e-16f);
  h2[i] += al * g[i];
}

// generic eval-BN (+optional pre-bias, optional ELU), in place; C must be pow2
__global__ void k_bn_act(float* __restrict__ h, const float* __restrict__ prebias,
                         const float* __restrict__ g, const float* __restrict__ b,
                         const float* __restrict__ m, const float* __restrict__ v,
                         int total, int C, int act) {
  int i = blockIdx.x * blockDim.x + threadIdx.x;
  if (i >= total) return;
  int c = i & (C - 1);
  float val = h[i] + (prebias ? prebias[c] : 0.f);
  val = (val - m[c]) * (g[c] * rsqrtf(v[c] + EPSBN)) + b[c];
  if (act) val = eluf(val);
  h[i] = val;
}

// wave-per-edge: mean[dst] += h2[src]
__global__ void k_sage_scatter(const float* __restrict__ h2, const int* __restrict__ src,
                               const int* __restrict__ dst, float* __restrict__ mean, int E) {
  int gw = (blockIdx.x * blockDim.x + threadIdx.x) >> 5;
  int nw = (gridDim.x * blockDim.x) >> 5;
  int lane = threadIdx.x & 31;
  for (int e = gw; e < E; e += nw) {
    int s = src[e], d = dst[e];
    const float* hs = h2 + (size_t)s * HIDD;
    float* md = mean + (size_t)d * HIDD;
    #pragma unroll
    for (int c = 0; c < 4; ++c)
      atomicAdd(&md[lane + c * 32], hs[lane + c * 32]);
  }
}

__global__ void k_sage_mean(float* __restrict__ mean, const int* __restrict__ cnt, int total) {
  int i = blockIdx.x * blockDim.x + threadIdx.x;
  if (i >= total) return;
  int node = i >> 7;
  int c = cnt[node];
  mean[i] *= (c > 0) ? (1.f / (float)c) : 1.f;
}

// ---------------- launcher ----------------
extern "C" void kernel_launch(void* const* d_in, const int* in_sizes, int n_in,
                              void* d_out, int out_size, void* d_ws, size_t ws_size,
                              hipStream_t stream) {
  const float* x       = (const float*)d_in[0];
  const int*   ei      = (const int*)d_in[1];
  const float* w_in    = (const float*)d_in[2];
  const float* b_in    = (const float*)d_in[3];
  const float* w_gcn   = (const float*)d_in[4];
  const float* b_gcn   = (const float*)d_in[5];
  const float* bn1_g   = (const float*)d_in[6];
  const float* bn1_b   = (const float*)d_in[7];
  const float* bn1_m   = (const float*)d_in[8];
  const float* bn1_v   = (const float*)d_in[9];
  const float* w_gat   = (const float*)d_in[10];
  const float* att_src = (const float*)d_in[11];
  const float* att_dst = (const float*)d_in[12];
  const float* b_gat   = (const float*)d_in[13];
  const float* bn2_g   = (const float*)d_in[14];
  const float* bn2_b   = (const float*)d_in[15];
  const float* bn2_m   = (const float*)d_in[16];
  const float* bn2_v   = (const float*)d_in[17];
  const float* w_sgl   = (const float*)d_in[18];
  const float* b_sage  = (const float*)d_in[19];
  const float* w_sgr   = (const float*)d_in[20];
  const float* bn3_g   = (const float*)d_in[21];
  const float* bn3_b   = (const float*)d_in[22];
  const float* bn3_m   = (const float*)d_in[23];
  const float* bn3_v   = (const float*)d_in[24];
  const float* w_res   = (const float*)d_in[25];
  const float* b_res   = (const float*)d_in[26];
  const float* w_c1    = (const float*)d_in[27];
  const float* b_c1    = (const float*)d_in[28];
  const float* w_c2    = (const float*)d_in[29];
  const float* b_c2    = (const float*)d_in[30];

  const int n = in_sizes[0] / F_IN;
  const int E = in_sizes[1] / 2;
  const int* src = ei;
  const int* dst = ei + E;

  // ---- workspace carve ----
  char* p = (char*)d_ws;
  auto carve = [&](size_t bytes) -> void* {
    void* r = (void*)p;
    p += (bytes + 255) & ~(size_t)255;
    return r;
  };
  float*    bufH  = (float*)carve((size_t)n * 128 * 4);  // h / residual
  float*    bufG  = (float*)carve((size_t)n * 128 * 4);  // hw -> g -> mean_nb
  float*    bufA  = (float*)carve((size_t)n * 128 * 4);  // h1 -> h2
  float*    bufT  = (float*)carve((size_t)n * 64 * 4);   // h3
  float*    bufZ  = (float*)carve((size_t)n * 32 * 4);   // z
  float*    a_s   = (float*)carve((size_t)n * 2 * 4);
  float*    a_d   = (float*)carve((size_t)n * 2 * 4);
  float*    den   = (float*)carve((size_t)n * 2 * 4);
  unsigned* mxu   = (unsigned*)carve((size_t)n * 2 * 4);
  int*      cnt   = (int*)carve((size_t)n * 4);
  float*    dinv  = (float*)carve((size_t)n * 4);
  _Float16* f_in  = (_Float16*)carve((size_t)128 * 192 * 2);  // transposed [Nout][Kp]
  _Float16* f_gcn = (_Float16*)carve((size_t)128 * 128 * 2);
  _Float16* f_gat = (_Float16*)carve((size_t)128 * 128 * 2);
  _Float16* f_sgl = (_Float16*)carve((size_t)64 * 128 * 2);
  _Float16* f_sgr = (_Float16*)carve((size_t)64 * 128 * 2);
  _Float16* f_res = (_Float16*)carve((size_t)64 * 128 * 2);
  _Float16* f_c1  = (_Float16*)carve((size_t)32 * 64 * 2);
  _Float16* f_c2  = (_Float16*)carve((size_t)16 * 32 * 2);

  const int TB = 256;
  dim3 blk(TB);
  dim3 gemm_grid((n + 127) / 128);
  dim3 ew128_grid((n * 128 + TB - 1) / TB);
  dim3 ew2_grid((n * 2 + TB - 1) / TB);
  dim3 node_grid((n + TB - 1) / TB);
  dim3 edge_grid((E + TB - 1) / TB);
  dim3 wave_grid(2048);

  auto conv = [&](const float* s, _Float16* d, int K, int Ns, int Kp, int Np) {
    int total = Kp * Np;
    hipLaunchKernelGGL(k_convert_wT, dim3((total + TB - 1) / TB), blk, 0, stream,
                       s, d, K, Ns, Kp, Np);
  };
  conv(w_in, f_in, 165, 128, 192, 128);
  conv(w_gcn, f_gcn, 128, 128, 128, 128);
  conv(w_gat, f_gat, 128, 128, 128, 128);
  conv(w_sgl, f_sgl, 128, 64, 128, 64);
  conv(w_sgr, f_sgr, 128, 64, 128, 64);
  conv(w_res, f_res, 128, 64, 128, 64);
  conv(w_c1, f_c1, 64, 32, 64, 32);
  conv(w_c2, f_c2, 32, 2, 32, 16);

  // degrees (cnt = in-degree without self loops; GCN uses cnt+1, SAGE uses cnt)
  hipLaunchKernelGGL(k_zero, dim3(1024), blk, 0, stream, (unsigned*)cnt, (long)n);
  hipLaunchKernelGGL(k_degree, edge_grid, blk, 0, stream, dst, E, cnt);
  hipLaunchKernelGGL(k_dinv, node_grid, blk, 0, stream, cnt, dinv, n);

  // h = elu(x @ w_in + b_in)
  hipLaunchKernelGGL((k_gemm_wmma<8>), gemm_grid, blk, 0, stream,
                     x, f_in, bufH, b_in, n, 165, 192, 128, 128, 0, 1);
  // hw = h @ w_gcn
  hipLaunchKernelGGL((k_gemm_wmma<8>), gemm_grid, blk, 0, stream,
                     bufH, f_gcn, bufG, (const float*)nullptr, n, 128, 128, 128, 128, 0, 0);
  // GCN aggregate
  hipLaunchKernelGGL(k_zero, dim3(4096), blk, 0, stream, (unsigned*)bufA, (long)n * 128);
  hipLaunchKernelGGL(k_gcn_scatter, wave_grid, blk, 0, stream, bufG, src, dst, dinv, bufA, E);
  hipLaunchKernelGGL(k_gcn_finish, ew128_grid, blk, 0, stream,
                     bufA, bufG, dinv, b_gcn, bn1_g, bn1_b, bn1_m, bn1_v, n * 128);
  // g = h1 @ w_gat
  hipLaunchKernelGGL((k_gemm_wmma<8>), gemm_grid, blk, 0, stream,
                     bufA, f_gat, bufG, (const float*)nullptr, n, 128, 128, 128, 128, 0, 0);
  // attention logits
  hipLaunchKernelGGL(k_gat_att, ew2_grid, blk, 0, stream, bufG, att_src, att_dst, a_s, a_d, n);
  // softmax: max
  hipLaunchKernelGGL(k_zero, dim3(1024), blk, 0, stream, mxu, (long)n * 2);
  hipLaunchKernelGGL(k_gat_edge_max, edge_grid, blk, 0, stream, src, dst, a_s, a_d, mxu, E);
  hipLaunchKernelGGL(k_gat_self_max, ew2_grid, blk, 0, stream, a_s, a_d, mxu, n * 2);
  // softmax: denom
  hipLaunchKernelGGL(k_zero, dim3(1024), blk, 0, stream, (unsigned*)den, (long)n * 2);
  hipLaunchKernelGGL(k_gat_edge_denom, edge_grid, blk, 0, stream, src, dst, a_s, a_d, mxu, den, E);
  hipLaunchKernelGGL(k_gat_self_denom, ew2_grid, blk, 0, stream, a_s, a_d, mxu, den, n * 2);
  // weighted aggregate -> h2 (reuse bufA)
  hipLaunchKernelGGL(k_zero, dim3(4096), blk, 0, stream, (unsigned*)bufA, (long)n * 128);
  hipLaunchKernelGGL(k_gat_edge_scatter, wave_grid, blk, 0, stream,
                     src, dst, a_s, a_d, mxu, den, bufG, bufA, E);
  hipLaunchKernelGGL(k_gat_self_scatter, ew128_grid, blk, 0, stream,
                     a_s, a_d, mxu, den, bufG, bufA, n * 128);
  // h2 = elu(bn2(h2 + b_gat))
  hipLaunchKernelGGL(k_bn_act, ew128_grid, blk, 0, stream,
                     bufA, b_gat, bn2_g, bn2_b, bn2_m, bn2_v, n * 128, 128, 1);
  // SAGE mean aggregate (reuse bufG)
  hipLaunchKernelGGL(k_zero, dim3(4096), blk, 0, stream, (unsigned*)bufG, (long)n * 128);
  hipLaunchKernelGGL(k_sage_scatter, wave_grid, blk, 0, stream, bufA, src, dst, bufG, E);
  hipLaunchKernelGGL(k_sage_mean, ew128_grid, blk, 0, stream, bufG, cnt, n * 128);
  // t = mean @ w_sage_l + b_sage ; t += h2 @ w_sage_r ; t = bn3(t)
  hipLaunchKernelGGL((k_gemm_wmma<4>), gemm_grid, blk, 0, stream,
                     bufG, f_sgl, bufT, b_sage, n, 128, 128, 64, 64, 0, 0);
  hipLaunchKernelGGL((k_gemm_wmma<4>), gemm_grid, blk, 0, stream,
                     bufA, f_sgr, bufT, (const float*)nullptr, n, 128, 128, 64, 64, 1, 0);
  hipLaunchKernelGGL(k_bn_act, dim3((n * 64 + TB - 1) / TB), blk, 0, stream,
                     bufT, (const float*)nullptr, bn3_g, bn3_b, bn3_m, bn3_v, n * 64, 64, 0);
  // h3 = elu(t + residual @ w_res + b_res)
  hipLaunchKernelGGL((k_gemm_wmma<4>), gemm_grid, blk, 0, stream,
                     bufH, f_res, bufT, b_res, n, 128, 128, 64, 64, 1, 1);
  // z = elu(h3 @ w_c1 + b_c1)
  hipLaunchKernelGGL((k_gemm_wmma<2>), gemm_grid, blk, 0, stream,
                     bufT, f_c1, bufZ, b_c1, n, 64, 64, 32, 32, 0, 1);
  // out = z @ w_c2 + b_c2  (Nout padded to 16, store only 2 cols, ldc=2)
  hipLaunchKernelGGL((k_gemm_wmma<1>), gemm_grid, blk, 0, stream,
                     bufZ, f_c2, (float*)d_out, b_c2, n, 32, 32, 2, 2, 0, 0);
}
